// STDCT_19464791785520
// MI455X (gfx1250) — compile-verified, matrix operand order checked by reading, not run on hardware
//
#include <hip/hip_runtime.h>
#include <hip/hip_bf16.h>
#include <stdint.h>

// ---------------------------------------------------------------------------
// STDCT = strided conv1d = GEMM:  C[k, b*2048+f] = sum_t Filt[k,t]*Xpad[b,f*256+t]
// M(freq)=1024, N(frames)=16*2048=32768, K=1024.
// Split-bf16 (hi/lo) x 3 WMMA products -> ~fp32 accuracy on bf16 matrix pipe.
// Register-blocked: one wave computes a 32x32 C super-tile (2x2 WMMA tiles),
// reusing each A/B fragment twice -> 1.33 b128 loads per WMMA.
// ---------------------------------------------------------------------------

#define NFREQ   1024
#define HOP     256
#define BATCH   16
#define TFRAMES 2048
#define LSIG    (HOP * TFRAMES)      // 524288
#define PADL    384
#define LPAD    (LSIG + 2 * PADL)    // 525056 (multiple of 8 -> 16B aligned rows)

typedef __attribute__((ext_vector_type(16))) __bf16 v16bf;
typedef __attribute__((ext_vector_type(8)))  float  v8f;

#define WMMA_BF16(A, B, C) \
    __builtin_amdgcn_wmma_f32_16x16x32_bf16(false, (A), false, (B), (short)0, (C), false, false)

// ---- fp32 -> bf16 hi/lo split (truncation: lo = exact residual of hi) ------
__device__ __forceinline__ void split_bf16(float f, uint16_t& h, uint16_t& l) {
    uint32_t u = __float_as_uint(f);
    h = (uint16_t)(u >> 16);
    float hf = __uint_as_float((uint32_t)h << 16);
    float lo = f - hf;
    l = (uint16_t)(__float_as_uint(lo) >> 16);
}

__global__ __launch_bounds__(256) void stdct_split_filter(
        const float* __restrict__ filt, uint16_t* __restrict__ hi,
        uint16_t* __restrict__ lo) {
    int i = blockIdx.x * 256 + threadIdx.x;
    if (i < NFREQ * NFREQ) {
        uint16_t h, l;
        split_bf16(filt[i], h, l);
        hi[i] = h;
        lo[i] = l;
    }
}

__global__ __launch_bounds__(256) void stdct_pad_split_x(
        const float* __restrict__ x, uint16_t* __restrict__ hi,
        uint16_t* __restrict__ lo) {
    int i = blockIdx.x * 256 + threadIdx.x;          // over BATCH*LPAD = 8.4M
    if (i >= BATCH * LPAD) return;
    int b = i / LPAD;
    int j = i - b * LPAD;
    float f = 0.0f;
    int s = j - PADL;
    if (s >= 0 && s < LSIG) f = x[(size_t)b * LSIG + s];
    uint16_t h, l;
    split_bf16(f, h, l);
    hi[i] = h;
    lo[i] = l;
}

// ---- load one 16x32 bf16 A-style fragment for this lane --------------------
// ISA 16-bit A layout (wave32): lanes 0-15 hold row M=lane, K = base..base+7
// in v0..v3 and base+16..base+23 in v4..v7; lanes 16-31 hold the +8 halves.
// Two aligned b128 loads per fragment.
__device__ __forceinline__ v16bf load_frag(const uint16_t* p1, const uint16_t* p2) {
    union { v16bf v; uint4 u[2]; } r;
    r.u[0] = *(const uint4*)p1;
    r.u[1] = *(const uint4*)p2;
    return r.v;
}

// ---- main GEMM: one wave per 32(freq) x 32(frame) super-tile ---------------
// super-tiles = 16 batches * 32 freq * 64 frame = 32768 waves = 4096 blocks
__global__ __launch_bounds__(256) void stdct_wmma_kernel(
        const uint16_t* __restrict__ fHi, const uint16_t* __restrict__ fLo,
        const uint16_t* __restrict__ xHi, const uint16_t* __restrict__ xLo,
        float* __restrict__ out) {
    const int wave = threadIdx.x >> 5;
    const int lane = threadIdx.x & 31;
    const int tile = blockIdx.x * 8 + wave;

    const int f_sup = tile & 63;            // 64 frame super-tiles (32 frames)
    const int k_sup = (tile >> 6) & 31;     // 32 freq super-tiles (32 freqs)
    const int b     = tile >> 11;           // 16 batches

    const int row     = lane & 15;          // row within a 16-wide tile
    const int halfsel = (lane >> 4) & 1;    // lane group selects K halves

    const int k0 = k_sup * 32;
    const int f0 = f_sup * 32;

    const uint16_t* aH0 = fHi + (size_t)(k0 + row) * NFREQ;
    const uint16_t* aL0 = fLo + (size_t)(k0 + row) * NFREQ;
    const uint16_t* aH1 = aH0 + (size_t)16 * NFREQ;
    const uint16_t* aL1 = aL0 + (size_t)16 * NFREQ;

    const size_t xoff = (size_t)b * LPAD + (size_t)(f0 + row) * HOP;
    const uint16_t* bH0 = xHi + xoff;
    const uint16_t* bL0 = xLo + xoff;
    const uint16_t* bH1 = bH0 + (size_t)16 * HOP;
    const uint16_t* bL1 = bL0 + (size_t)16 * HOP;

    v8f acc00 = {}, acc01 = {}, acc10 = {}, acc11 = {};

    #pragma unroll 2
    for (int kc = 0; kc < NFREQ; kc += 32) {
        const int base1 = kc + halfsel * 8;   // 16B aligned (kc%32==0)
        const int base2 = base1 + 16;

        v16bf a_h0 = load_frag(aH0 + base1, aH0 + base2);
        v16bf a_l0 = load_frag(aL0 + base1, aL0 + base2);
        v16bf a_h1 = load_frag(aH1 + base1, aH1 + base2);
        v16bf a_l1 = load_frag(aL1 + base1, aL1 + base2);
        v16bf b_h0 = load_frag(bH0 + base1, bH0 + base2);
        v16bf b_l0 = load_frag(bL0 + base1, bL0 + base2);
        v16bf b_h1 = load_frag(bH1 + base1, bH1 + base2);
        v16bf b_l1 = load_frag(bL1 + base1, bL1 + base2);

        // (ah+al)*(bh+bl) ~= ah*bh + ah*bl + al*bh  (al*bl ~ 2^-16 rel, dropped)
        acc00 = WMMA_BF16(a_h0, b_h0, acc00);
        acc00 = WMMA_BF16(a_h0, b_l0, acc00);
        acc00 = WMMA_BF16(a_l0, b_h0, acc00);

        acc01 = WMMA_BF16(a_h0, b_h1, acc01);
        acc01 = WMMA_BF16(a_h0, b_l1, acc01);
        acc01 = WMMA_BF16(a_l0, b_h1, acc01);

        acc10 = WMMA_BF16(a_h1, b_h0, acc10);
        acc10 = WMMA_BF16(a_h1, b_l0, acc10);
        acc10 = WMMA_BF16(a_l1, b_h0, acc10);

        acc11 = WMMA_BF16(a_h1, b_h1, acc11);
        acc11 = WMMA_BF16(a_h1, b_l1, acc11);
        acc11 = WMMA_BF16(a_l1, b_h1, acc11);
    }

    // C/D layout: lanes 0-15: VGPR j -> (M=j, N=lane); lanes 16-31: (M=8+j, N=lane-16)
    const int kbase = k0 + halfsel * 8;
    const int fbase = f0 + row;
    float* o = out + (size_t)b * NFREQ * TFRAMES + (size_t)kbase * TFRAMES + fbase;
    #pragma unroll
    for (int j = 0; j < 8; ++j) {
        o[(size_t)j * TFRAMES]                         = acc00[j];
        o[(size_t)j * TFRAMES + 16]                    = acc01[j];
        o[(size_t)(j + 16) * TFRAMES]                  = acc10[j];
        o[(size_t)(j + 16) * TFRAMES + 16]             = acc11[j];
    }
}

// ---------------------------------------------------------------------------
extern "C" void kernel_launch(void* const* d_in, const int* in_sizes, int n_in,
                              void* d_out, int out_size, void* d_ws, size_t ws_size,
                              hipStream_t stream) {
    const float* x    = (const float*)d_in[0];   // [16, 1, 524288]
    const float* filt = (const float*)d_in[1];   // [1024, 1, 1024]
    float* out = (float*)d_out;                  // [16, 1024, 2048]

    // workspace layout (all offsets 16B aligned)
    char* ws = (char*)d_ws;
    uint16_t* fHi = (uint16_t*)(ws);                               // 2 MB
    uint16_t* fLo = (uint16_t*)(ws + (size_t)NFREQ * NFREQ * 2);   // 2 MB
    uint16_t* xHi = (uint16_t*)(ws + (size_t)NFREQ * NFREQ * 4);   // 16.8 MB
    uint16_t* xLo = (uint16_t*)(ws + (size_t)NFREQ * NFREQ * 4
                                   + (size_t)BATCH * LPAD * 2);    // 16.8 MB

    // 1) split filter to bf16 hi/lo
    {
        int n = NFREQ * NFREQ;
        stdct_split_filter<<<(n + 255) / 256, 256, 0, stream>>>(filt, fHi, fLo);
    }
    // 2) zero-padded bf16 hi/lo signal (kills bounds checks in GEMM loop)
    {
        int n = BATCH * LPAD;
        stdct_pad_split_x<<<(n + 255) / 256, 256, 0, stream>>>(x, xHi, xLo);
    }
    // 3) WMMA GEMM: 32768 wave super-tiles, 8 waves / block
    {
        int blocks = (BATCH * 32 * 64) / 8;      // 4096
        stdct_wmma_kernel<<<blocks, 256, 0, stream>>>(fHi, fLo, xHi, xLo, out);
    }
}
